// GINEncoder_76897094468164
// MI455X (gfx1250) — matrix operand order repaired and checked by reading, not compiled, not run on hardware
//
#include <hip/hip_runtime.h>
#include <hip/hip_bf16.h>

// ---------------------------------------------------------------------------
// GIN encoder for MI455X (gfx1250, wave32).
//   per layer: agg = h + scatter_add(h[src]->dst)
//              z   = agg @ W1 + b1             (WMMA f16 -> f32)
//              t   = relu(BN1(z))              (fused into GEMM2 A-load)
//              u   = relu(t @ W2 + b2)         (WMMA f16 -> f32)
//              h   = BN2(u)                    -> node_h slab + next layer
//   graph_h computed analytically from BN2 stats: scale*colsum + N*shift.
// ---------------------------------------------------------------------------

typedef __attribute__((ext_vector_type(16))) _Float16 v16h;
typedef __attribute__((ext_vector_type(8)))  _Float16 v8h;
typedef __attribute__((ext_vector_type(8)))  float    v8f;

#define DIMS      64
#define WT_STRIDE 72   // LDS row stride in halves (144B) to stagger banks
#define BN_EPS    1e-5f

// ---------------------------- small utility kernels ------------------------

__global__ __launch_bounds__(256) void zero_kernel(float* __restrict__ p, int n) {
    int i = blockIdx.x * 256 + threadIdx.x;
    if (i < n) p[i] = 0.0f;
}

__global__ __launch_bounds__(256) void copy_f4_kernel(const float* __restrict__ s,
                                                      float* __restrict__ d,
                                                      long long n4) {
    long long i = (long long)blockIdx.x * 256 + threadIdx.x;
    if (i < n4) ((float4*)d)[i] = ((const float4*)s)[i];
}

// One wave per edge; each lane moves 2 floats (float2 load, 2 f32 atomics).
// Feature matrix is 25.6 MB -> resident in the 192 MB L2, so both the random
// row gather and the atomic adds stay on-chip.
__global__ __launch_bounds__(256) void scatter_add_kernel(
    const float* __restrict__ H, const int* __restrict__ src,
    const int* __restrict__ dst, float* __restrict__ Agg, int E) {
    int tid = blockIdx.x * 256 + threadIdx.x;
    int e = tid >> 5;
    if (e >= E) return;
    int c = (tid & 31) * 2;
    long long s = src[e], d = dst[e];
    float2 v = *(const float2*)(H + s * DIMS + c);
    atomicAdd(Agg + d * DIMS + c,     v.x);
    atomicAdd(Agg + d * DIMS + c + 1, v.y);
}

// -------------------------- fused WMMA GEMM + BN stats ----------------------
// Block = 256 thr = 8 waves; wave computes a 16x64 output stripe:
//   4 N-tiles x 2 K-chunks of v_wmma_f32_16x16x32_f16.
// APPLY_BN: fuse BN(scale,shift)+ReLU into the A f32->f16 convert.
// OUT_RELU: ReLU the output before store/stats.
// BN stats: lane partials -> LDS ds_add_f32 -> 128 global atomics per block.
// Tail handling is wave-uniform: only the last block takes the guarded path,
// so the hot epilogue is straight-line immediate-offset stores (EXEC all-1s
// at every WMMA either way).
template <bool APPLY_BN, bool OUT_RELU>
__global__ __launch_bounds__(256) void gemm_bn_stats_kernel(
    const float* __restrict__ A,        // [N,64]
    const float* __restrict__ W,        // [64,64] row-major (k,n)
    const float* __restrict__ bias,     // [64]
    const float* __restrict__ bnScale,  // [64] (used iff APPLY_BN)
    const float* __restrict__ bnShift,  // [64]
    float* __restrict__ Out,            // [N,64]
    float* __restrict__ statSum,        // [64]
    float* __restrict__ statSq,         // [64]
    int N) {

    __shared__ _Float16 Wt[DIMS * WT_STRIDE];  // transposed weights, f16
    __shared__ float sSum[DIMS], sSq[DIMS];

    const int tid  = threadIdx.x;
    const int wave = tid >> 5;
    const int lane = tid & 31;

    if (tid < DIMS) { sSum[tid] = 0.0f; sSq[tid] = 0.0f; }

    // Stage weights into LDS transposed (coalesced global read).
    for (int idx = tid; idx < DIMS * DIMS; idx += 256) {
        int k = idx >> 6, n = idx & 63;
        Wt[n * WT_STRIDE + k] = (_Float16)W[idx];
    }
    __syncthreads();

    const int rowBase = blockIdx.x * 128 + wave * 16;
    const bool fullTile = (rowBase + 16 <= N);   // wave-uniform
    int arow = rowBase + (lane & 15);
    if (arow >= N) arow = N - 1;                 // clamp loads; stores masked

    // A fragments per ISA 16-bit A layout:
    //   lane<16 : K in {0..7, 16..23};  lane>=16 : K in {8..15, 24..31}
    const int halfSel = (lane < 16) ? 0 : 8;
    const float* arowp = A + (long long)arow * DIMS;
    v16h afrag[2];
#pragma unroll
    for (int kc = 0; kc < 2; ++kc) {
        const int c0 = kc * 32 + halfSel;        // K base of elems 0..7
        const int c1 = c0 + 16;                  // K base of elems 8..15
        float4 x0 = *(const float4*)(arowp + c0);
        float4 x1 = *(const float4*)(arowp + c0 + 4);
        float4 y0 = *(const float4*)(arowp + c1);
        float4 y1 = *(const float4*)(arowp + c1 + 4);
        if (APPLY_BN) {
            float4 s0 = *(const float4*)(bnScale + c0);
            float4 s1 = *(const float4*)(bnScale + c0 + 4);
            float4 t0 = *(const float4*)(bnShift + c0);
            float4 t1 = *(const float4*)(bnShift + c0 + 4);
            float4 u0 = *(const float4*)(bnScale + c1);
            float4 u1 = *(const float4*)(bnScale + c1 + 4);
            float4 w0 = *(const float4*)(bnShift + c1);
            float4 w1 = *(const float4*)(bnShift + c1 + 4);
            x0.x = fmaxf(x0.x * s0.x + t0.x, 0.f); x0.y = fmaxf(x0.y * s0.y + t0.y, 0.f);
            x0.z = fmaxf(x0.z * s0.z + t0.z, 0.f); x0.w = fmaxf(x0.w * s0.w + t0.w, 0.f);
            x1.x = fmaxf(x1.x * s1.x + t1.x, 0.f); x1.y = fmaxf(x1.y * s1.y + t1.y, 0.f);
            x1.z = fmaxf(x1.z * s1.z + t1.z, 0.f); x1.w = fmaxf(x1.w * s1.w + t1.w, 0.f);
            y0.x = fmaxf(y0.x * u0.x + w0.x, 0.f); y0.y = fmaxf(y0.y * u0.y + w0.y, 0.f);
            y0.z = fmaxf(y0.z * u0.z + w0.z, 0.f); y0.w = fmaxf(y0.w * u0.w + w0.w, 0.f);
            y1.x = fmaxf(y1.x * u1.x + w1.x, 0.f); y1.y = fmaxf(y1.y * u1.y + w1.y, 0.f);
            y1.z = fmaxf(y1.z * u1.z + w1.z, 0.f); y1.w = fmaxf(y1.w * u1.w + w1.w, 0.f);
        }
        afrag[kc][0]  = (_Float16)x0.x; afrag[kc][1]  = (_Float16)x0.y;
        afrag[kc][2]  = (_Float16)x0.z; afrag[kc][3]  = (_Float16)x0.w;
        afrag[kc][4]  = (_Float16)x1.x; afrag[kc][5]  = (_Float16)x1.y;
        afrag[kc][6]  = (_Float16)x1.z; afrag[kc][7]  = (_Float16)x1.w;
        afrag[kc][8]  = (_Float16)y0.x; afrag[kc][9]  = (_Float16)y0.y;
        afrag[kc][10] = (_Float16)y0.z; afrag[kc][11] = (_Float16)y0.w;
        afrag[kc][12] = (_Float16)y1.x; afrag[kc][13] = (_Float16)y1.y;
        afrag[kc][14] = (_Float16)y1.z; afrag[kc][15] = (_Float16)y1.w;
    }

    const int colInTile = lane & 15;
    const int rowLane   = rowBase + ((lane < 16) ? 0 : 8);  // first row this lane holds
#pragma unroll
    for (int nt = 0; nt < 4; ++nt) {
        const int col = nt * 16 + colInTile;
        float bv = bias[col];
        v8f c = {bv, bv, bv, bv, bv, bv, bv, bv};  // bias pre-loaded into C
#pragma unroll
        for (int kc = 0; kc < 2; ++kc) {
            // B fragment: lanes 0-15 hold K=kc*32+{0..7,16..23}, lanes 16-31 +{8..15,24..31}
            const _Float16* bp =
                &Wt[col * WT_STRIDE + kc * 32 + ((lane < 16) ? 0 : 8)];
            union { v16h v; v8h h[2]; } bu;
            bu.h[0] = *(const v8h*)(bp);
            bu.h[1] = *(const v8h*)(bp + 16);
            c = __builtin_amdgcn_wmma_f32_16x16x32_f16(
                    false, afrag[kc], false, bu.v, (short)0, c, false, false);
        }
        // D layout: VGPR r -> row rowLane + r, column = col.
        float ps = 0.0f, pq = 0.0f;
        float* outp = Out + (long long)rowLane * DIMS + col;
        if (fullTile) {
#pragma unroll
            for (int r = 0; r < 8; ++r) {
                float v = c[r];
                if (OUT_RELU) v = fmaxf(v, 0.0f);
                outp[r * DIMS] = v;                 // imm offsets, +256B/row
                ps += v;
                pq += v * v;
            }
        } else {
#pragma unroll
            for (int r = 0; r < 8; ++r) {
                float v = c[r];
                if (OUT_RELU) v = fmaxf(v, 0.0f);
                if (rowLane + r < N) {
                    outp[r * DIMS] = v;
                    ps += v;
                    pq += v * v;
                }
            }
        }
        atomicAdd(&sSum[col], ps);
        atomicAdd(&sSq[col],  pq);
    }
    __syncthreads();
    if (tid < DIMS) {
        atomicAdd(&statSum[tid], sSum[tid]);
        atomicAdd(&statSq[tid],  sSq[tid]);
    }
}

// scale = gamma * rsqrt(var+eps); shift = beta - mean*scale.
// For the outer BN also emits graph_h column = scale*colsum + N*shift.
__global__ void bn_finalize_kernel(const float* __restrict__ statSum,
                                   const float* __restrict__ statSq,
                                   const float* __restrict__ gamma,
                                   const float* __restrict__ beta,
                                   float* __restrict__ scale,
                                   float* __restrict__ shift,
                                   float* __restrict__ graphOut,  // or nullptr
                                   float invN, float fN) {
    int c = threadIdx.x;  // 64 threads
    float s  = statSum[c];
    float m  = s * invN;
    float v  = statSq[c] * invN - m * m;
    float sc = gamma[c] * rsqrtf(v + BN_EPS);
    float sh = beta[c] - m * sc;
    scale[c] = sc;
    shift[c] = sh;
    if (graphOut) graphOut[c] = sc * s + fN * sh;
}

// h = t*scale + shift; write both the next-layer buffer ([N,64] contiguous)
// and the node_h slab in d_out (row stride 192 floats).
__global__ __launch_bounds__(256) void bn_apply_kernel(
    const float* __restrict__ T, const float* __restrict__ scale,
    const float* __restrict__ shift, float* __restrict__ Hnext,
    float* __restrict__ nodeOut, int N) {
    int tid = blockIdx.x * 256 + threadIdx.x;
    int n  = tid >> 4;
    int c4 = (tid & 15) * 4;
    if (n >= N) return;
    float4 t = *(const float4*)(T + (long long)n * DIMS + c4);
    float4 o;
    o.x = t.x * scale[c4 + 0] + shift[c4 + 0];
    o.y = t.y * scale[c4 + 1] + shift[c4 + 1];
    o.z = t.z * scale[c4 + 2] + shift[c4 + 2];
    o.w = t.w * scale[c4 + 3] + shift[c4 + 3];
    *(float4*)(Hnext + (long long)n * DIMS + c4) = o;
    *(float4*)(nodeOut + (long long)n * (3 * DIMS) + c4) = o;
}

// ------------------------------- host driver -------------------------------

extern "C" void kernel_launch(void* const* d_in, const int* in_sizes, int n_in,
                              void* d_out, int out_size, void* d_ws, size_t ws_size,
                              hipStream_t stream) {
    const float* feats = (const float*)d_in[0];
    const int*   src   = (const int*)d_in[1];
    const int*   dst   = (const int*)d_in[2];
    const float* W1    = (const float*)d_in[3];
    const float* b1    = (const float*)d_in[4];
    const float* g1    = (const float*)d_in[5];
    const float* be1   = (const float*)d_in[6];
    const float* W2    = (const float*)d_in[7];
    const float* b2    = (const float*)d_in[8];
    const float* g2    = (const float*)d_in[9];
    const float* be2   = (const float*)d_in[10];

    const int N = in_sizes[0] / DIMS;
    const int E = in_sizes[1];
    const int L = 3;
    float* out = (float*)d_out;

    const long long nd = (long long)N * DIMS;
    float* bufA  = (float*)d_ws;       // h_next (layer input for i>0)
    float* bufB  = bufA + nd;          // agg, then t2
    float* bufC  = bufB + nd;          // z
    float* stats = bufC + nd;          // [2][4][64]: sum,sq,scale,shift x2
    float* sum0 = stats,       * sq0 = stats + 64;
    float* sc0  = stats + 128, * sh0 = stats + 192;
    float* sum1 = stats + 256, * sq1 = stats + 320;
    float* sc1  = stats + 384, * sh1 = stats + 448;

    const int copyBlocks    = (int)((nd / 4 + 255) / 256);
    const int scatterBlocks = (int)(((long long)E * 32 + 255) / 256);
    const int gemmBlocks    = (N + 127) / 128;
    const int applyBlocks   = (int)(((long long)N * 16 + 255) / 256);
    const float invN = 1.0f / (float)N;
    const float fN   = (float)N;

    const float* h = feats;
    for (int i = 0; i < L; ++i) {
        // zero BN accumulators (sum0,sq0,sc0,sh0 = 256 floats; sum1,sq1 not zeroed)
        zero_kernel<<<1, 256, 0, stream>>>(stats, 256);

        // agg = h ; agg += scatter(h[src] -> dst)
        copy_f4_kernel<<<copyBlocks, 256, 0, stream>>>(h, bufB, nd / 4);
        scatter_add_kernel<<<scatterBlocks, 256, 0, stream>>>(h, src, dst, bufB, E);

        // z = agg @ W1 + b1 ; BN1 stats
        gemm_bn_stats_kernel<false, false><<<gemmBlocks, 256, 0, stream>>>(
            bufB, W1 + (long long)i * DIMS * DIMS, b1 + i * DIMS,
            nullptr, nullptr, bufC, sum0, sq0, N);
        bn_finalize_kernel<<<1, 64, 0, stream>>>(
            sum0, sq0, g1 + i * DIMS, be1 + i * DIMS, sc0, sh0,
            nullptr, invN, fN);

        // t2 = relu( relu(BN1(z)) @ W2 + b2 ) ; BN2 stats
        gemm_bn_stats_kernel<true, true><<<gemmBlocks, 256, 0, stream>>>(
            bufC, W2 + (long long)i * DIMS * DIMS, b2 + i * DIMS,
            sc0, sh0, bufB, sum1, sq1, N);
        bn_finalize_kernel<<<1, 64, 0, stream>>>(
            sum1, sq1, g2 + i * DIMS, be2 + i * DIMS, sc1, sh1,
            out + i * DIMS, invN, fN);   // graph_h slab for this layer

        // h_next = BN2(t2) -> bufA and node_h slab of d_out
        bn_apply_kernel<<<applyBlocks, 256, 0, stream>>>(
            bufB, sc1, sh1, bufA, out + 3 * DIMS + (long long)i * DIMS, N);

        h = bufA;
    }
}
